// LancetBlock_30167850287558
// MI455X (gfx1250) — matrix operand order repaired and checked by reading, not compiled, not run on hardware
//
#include <hip/hip_runtime.h>
#include <hip/hip_bf16.h>

// ---------------------------------------------------------------------------
// Problem: B=8,S=8192 -> N=65536 tokens, D=512, H=4D=2048, E=8, 2 micro-batch
// chunks of 4096 tokens per (mb, expert). gate logits / top_k are dead code.
// Pipeline: LN(bf16) -> attn GEMM(+bias+residual) -> fc1(+bias,erf-GELU)
//           -> fc2(+bias) -> next(+bias,erf-GELU) -> fp32 out
// All GEMMs: bf16 inputs, f32 accum via v_wmma_f32_16x16x32_bf16.
// Tile staging: Tensor Data Mover (tensor_load_to_lds), double-buffered,
// TENSORcnt-synchronized; fallback to manual global->LDS copies if the TDM
// builtin is unavailable.
// ---------------------------------------------------------------------------
#define D_MODEL 512
#define D_FF 2048
#define N_TOK 65536
#define N_EXP 8
#define CHUNKS 16
#define CHUNK_ROWS 4096

typedef unsigned short u16;
typedef unsigned int u32;
typedef unsigned long long u64;
typedef __attribute__((ext_vector_type(16))) __bf16 v16bf;
typedef __attribute__((ext_vector_type(8))) float v8f;
typedef __attribute__((ext_vector_type(4))) unsigned int v4u;
typedef __attribute__((ext_vector_type(8))) int v8i;
typedef __attribute__((ext_vector_type(4))) int v4i;

#if defined(__gfx1250__) && __has_builtin(__builtin_amdgcn_tensor_load_to_lds)
#define USE_TDM 1
#else
#define USE_TDM 0
#endif

__device__ __forceinline__ u16 f32_to_bf16(float f) {
  u32 u = __float_as_uint(f);
  u32 r = (u + 0x7FFFu + ((u >> 16) & 1u)) >> 16;  // RNE
  return (u16)r;
}

__device__ __forceinline__ float gelu_erf(float x) {
  return 0.5f * x * (1.0f + erff(x * 0.70710678118654752440f));
}

#if USE_TDM
// Issue one TDM 2D tile load: tile_dim0=32 bf16 (one K-slab row segment),
// tile_dim1=128 rows, row stride = K elements. LDS padding: after every
// 16 DWORDs (32 bf16) insert 4 DWORDs -> padded LDS row stride of 40 bf16,
// matching the compute-side LDT without any VALU work.
__device__ __forceinline__ void tdm_load_tile(unsigned lds_byte_addr,
                                              const u16* tile, int K) {
  const u64 ga = (u64)tile;
  v4u g0;
  g0[0] = 1u;                              // count=1, user descriptor
  g0[1] = lds_byte_addr;                   // LDS byte address
  g0[2] = (u32)ga;                         // global_addr[31:0]
  g0[3] = (u32)(ga >> 32) | (2u << 30);    // global_addr[56:32] | type=2
  v8i g1;
  // data_size=1 (2B) | pad_enable | pad_interval=3 (16 DW) | pad_amount=3 (4 DW)
  g1[0] = (int)((1u << 16) | (1u << 20) | (3u << 22) | (3u << 25));
  g1[1] = (int)(((u32)K & 0xFFFFu) << 16);          // tensor_dim0[15:0]
  g1[2] = (int)(((u32)K >> 16) & 0xFFFFu);          // tensor_dim0[31:16], dim1 lo=0
  g1[3] = (int)(0x10u | (32u << 16));               // tensor_dim1=1M rows, tile_dim0=32
  g1[4] = (int)128u;                                // tile_dim1=128, tile_dim2=0
  g1[5] = (int)(u32)K;                              // tensor_dim0_stride[31:0]
  g1[6] = 0;                                        // stride hi / dim1_stride lo
  g1[7] = 0;
  v4i g2 = {};  // dims 2..4 unused (2D tile)
  v4i g3 = {};
#if __clang_major__ >= 23
  v8i g4 = {};
  __builtin_amdgcn_tensor_load_to_lds(g0, g1, g2, g3, g4, 0);
#else
  __builtin_amdgcn_tensor_load_to_lds(g0, g1, g2, g3, 0);
#endif
}
#endif

// ---------------------------------------------------------------------------
// fp32 -> bf16 weight conversion, 4-wide
// ---------------------------------------------------------------------------
__global__ void cvt_f32_bf16_x4(const float* __restrict__ in,
                                u16* __restrict__ out, int n4) {
  int i = blockIdx.x * blockDim.x + threadIdx.x;
  int stride = gridDim.x * blockDim.x;
  for (; i < n4; i += stride) {
    float4 f = ((const float4*)in)[i];
    ushort4 o;
    o.x = f32_to_bf16(f.x);
    o.y = f32_to_bf16(f.y);
    o.z = f32_to_bf16(f.z);
    o.w = f32_to_bf16(f.w);
    ((ushort4*)out)[i] = o;
  }
}

// ---------------------------------------------------------------------------
// LayerNorm over D=512, one wave32 per token (16 floats per lane), bf16 out
// ---------------------------------------------------------------------------
__launch_bounds__(256) __global__
void ln_to_bf16(const float* __restrict__ x, const float* __restrict__ g,
                const float* __restrict__ b, u16* __restrict__ out) {
  const int gtid = blockIdx.x * blockDim.x + threadIdx.x;
  const int tok = gtid >> 5;
  const int lane = gtid & 31;
  const float* row = x + (size_t)tok * D_MODEL;
  const int c0 = lane * 16;

  float4 v[4];
#pragma unroll
  for (int i = 0; i < 4; ++i) v[i] = *(const float4*)&row[c0 + i * 4];

  float sum = 0.f, sq = 0.f;
#pragma unroll
  for (int i = 0; i < 4; ++i) {
#pragma unroll
    for (int c = 0; c < 4; ++c) {
      float f = (&v[i].x)[c];
      sum += f;
      sq += f * f;
    }
  }
#pragma unroll
  for (int off = 16; off >= 1; off >>= 1) {
    sum += __shfl_xor(sum, off, 32);
    sq += __shfl_xor(sq, off, 32);
  }
  const float mu = sum * (1.0f / (float)D_MODEL);
  const float var = sq * (1.0f / (float)D_MODEL) - mu * mu;  // ddof=0
  const float rs = rsqrtf(var + 1e-5f);

  u16* orow = out + (size_t)tok * D_MODEL;
#pragma unroll
  for (int i = 0; i < 4; ++i) {
#pragma unroll
    for (int c = 0; c < 4; ++c) {
      int col = c0 + i * 4 + c;
      float f = ((&v[i].x)[c] - mu) * rs * g[col] + b[col];
      orow[col] = f32_to_bf16(f);
    }
  }
}

// ---------------------------------------------------------------------------
// Tiled bf16 WMMA GEMM:  out[m,n] = sum_k A[m,k] * B[n,k]  (+ epilogue)
// Block tile 128x128, 8 waves (4x2), wave tile 32x64 = 2x4 WMMA tiles.
// K staged in 32-wide slabs; A/B tiles moved to LDS by TDM, double-buffered.
// Epilogue modes: 0 +bias+resid->bf16 | 1 gelu(+bias)->bf16
//                 2 +bias->bf16       | 3 gelu(+bias)->f32
// ---------------------------------------------------------------------------
__launch_bounds__(256, 2) __global__
void gemm_bf16_wmma(const u16* __restrict__ A, const u16* __restrict__ B,
                    const float* __restrict__ bias,
                    const float* __restrict__ resid, u16* __restrict__ outB,
                    float* __restrict__ outF, int K, int N, int chunkRows,
                    int expertMask, int mode) {
  constexpr int LDT = 40;  // padded LDS row stride (bf16 elements)
  __shared__ u16 sA[2][128 * LDT];
  __shared__ u16 sB[2][128 * LDT];

  const int tid = threadIdx.x;
  const int lane = tid & 31;
  const int wid = tid >> 5;
  const int wm = wid & 3;
  const int wn = wid >> 2;
  const int half = lane >> 4;
  const int r = lane & 15;

  const int e = blockIdx.z & expertMask;
  const size_t rowBase = (size_t)blockIdx.z * chunkRows + (size_t)blockIdx.y * 128;
  const int colBase = blockIdx.x * 128;

  const u16* Ab = A + rowBase * (size_t)K;
  const u16* Bb = B + (size_t)e * N * K + (size_t)colBase * K;
  const float* biasE = bias + (size_t)e * N;

  v8f zero = {};
  v8f acc[2][4];
#pragma unroll
  for (int i = 0; i < 2; ++i)
#pragma unroll
    for (int j = 0; j < 4; ++j) acc[i][j] = zero;

  const int kTiles = K >> 5;

#if USE_TDM
  const unsigned ldsA[2] = {(unsigned)(u64)(const void*)&sA[0][0],
                            (unsigned)(u64)(const void*)&sA[1][0]};
  const unsigned ldsB[2] = {(unsigned)(u64)(const void*)&sB[0][0],
                            (unsigned)(u64)(const void*)&sB[1][0]};
  if (wid == 0) {  // one wave drives the Tensor Data Mover
    tdm_load_tile(ldsA[0], Ab, K);
    tdm_load_tile(ldsB[0], Bb, K);
  }
#endif

  for (int kt = 0; kt < kTiles; ++kt) {
    const int cur = kt & 1;
    const int k0 = kt << 5;
#if USE_TDM
    if (wid == 0) __builtin_amdgcn_s_wait_tensorcnt(0);  // buf[cur] landed
    __syncthreads();  // publish buf[cur]; all frag reads of buf[1-cur] done
    if (wid == 0 && kt + 1 < kTiles) {  // prefetch next slab into buf[1-cur]
      const int kn = (kt + 1) << 5;
      tdm_load_tile(ldsA[1 - cur], Ab + kn, K);
      tdm_load_tile(ldsB[1 - cur], Bb + kn, K);
    }
#else
    __syncthreads();
#pragma unroll
    for (int t = 0; t < 2; ++t) {
      int s = tid + t * 256;
      int row = s >> 2;
      int cs = (s & 3) << 3;
      *(uint4*)&sA[cur][row * LDT + cs] =
          *(const uint4*)&Ab[(size_t)row * K + k0 + cs];
      *(uint4*)&sB[cur][row * LDT + cs] =
          *(const uint4*)&Bb[(size_t)row * K + k0 + cs];
    }
    __syncthreads();
#endif
    (void)k0;

    union V {
      uint4 q[2];
      v16bf v;
    };
    V afr[2], bfr[4];
    // A 16x32 layout: lane(half,r) holds K = half*8+[0..7] and 16+half*8+[0..7]
#pragma unroll
    for (int i = 0; i < 2; ++i) {
      const u16* p = &sA[cur][(wm * 32 + i * 16 + r) * LDT + half * 8];
      afr[i].q[0] = *(const uint4*)p;
      afr[i].q[1] = *(const uint4*)(p + 16);
    }
    // B 32x16 layout: lane(half,r)=col r holds K = half*8+[0..7] and 16+half*8+[0..7]
#pragma unroll
    for (int j = 0; j < 4; ++j) {
      const u16* p = &sB[cur][(wn * 64 + j * 16 + r) * LDT + half * 8];
      bfr[j].q[0] = *(const uint4*)p;
      bfr[j].q[1] = *(const uint4*)(p + 16);
    }
#pragma unroll
    for (int i = 0; i < 2; ++i)
#pragma unroll
      for (int j = 0; j < 4; ++j)
        acc[i][j] = __builtin_amdgcn_wmma_f32_16x16x32_bf16(
            false, afr[i].v, false, bfr[j].v, (short)0, acc[i][j], false, false);
  }

  // Epilogue. C layout: VGPR v of a 16x16 tile -> row (v + 8*half), col r.
#pragma unroll
  for (int i = 0; i < 2; ++i) {
#pragma unroll
    for (int j = 0; j < 4; ++j) {
      const int ncol = colBase + wn * 64 + j * 16 + r;
      const float bv = biasE[ncol];
#pragma unroll
      for (int v = 0; v < 8; ++v) {
        const int mrow = wm * 32 + i * 16 + v + half * 8;
        const size_t oidx = (rowBase + mrow) * (size_t)N + ncol;
        float val = acc[i][j][v] + bv;
        if (mode == 0) {
          val += resid[oidx];
          outB[oidx] = f32_to_bf16(val);
        } else if (mode == 1) {
          outB[oidx] = f32_to_bf16(gelu_erf(val));
        } else if (mode == 2) {
          outB[oidx] = f32_to_bf16(val);
        } else {
          outF[oidx] = gelu_erf(val);
        }
      }
    }
  }
}

// ---------------------------------------------------------------------------
extern "C" void kernel_launch(void* const* d_in, const int* in_sizes, int n_in,
                              void* d_out, int out_size, void* d_ws,
                              size_t ws_size, hipStream_t stream) {
  const float* x = (const float*)d_in[0];
  const float* ln_g = (const float*)d_in[1];
  const float* ln_b = (const float*)d_in[2];
  const float* attn_w = (const float*)d_in[3];
  const float* attn_b = (const float*)d_in[4];
  /* d_in[5] = gate_w: dead in the reference output path */
  const float* fc1_w = (const float*)d_in[6];
  const float* fc1_b = (const float*)d_in[7];
  const float* fc2_w = (const float*)d_in[8];
  const float* fc2_b = (const float*)d_in[9];
  const float* next_w = (const float*)d_in[10];
  const float* next_b = (const float*)d_in[11];
  float* out = (float*)d_out;

  char* ws = (char*)d_ws;
  size_t off = 0;
  auto alloc = [&](size_t bytes) -> char* {
    char* p = ws + off;
    off += (bytes + 255) & ~(size_t)255;
    return p;
  };
  u16* normed = (u16*)alloc((size_t)N_TOK * D_MODEL * 2);
  u16* xattn = (u16*)alloc((size_t)N_TOK * D_MODEL * 2);
  u16* hbuf = (u16*)alloc((size_t)N_TOK * D_FF * 2);
  u16* ybuf = (u16*)alloc((size_t)N_TOK * D_MODEL * 2);
  u16* attn_wb = (u16*)alloc((size_t)D_MODEL * D_MODEL * 2);
  u16* fc1_wb = (u16*)alloc((size_t)N_EXP * D_FF * D_MODEL * 2);
  u16* fc2_wb = (u16*)alloc((size_t)N_EXP * D_MODEL * D_FF * 2);
  u16* next_wb = (u16*)alloc((size_t)D_MODEL * D_MODEL * 2);

  cvt_f32_bf16_x4<<<128, 256, 0, stream>>>(attn_w, attn_wb,
                                           D_MODEL * D_MODEL / 4);
  cvt_f32_bf16_x4<<<1024, 256, 0, stream>>>(fc1_w, fc1_wb,
                                            N_EXP * D_FF * D_MODEL / 4);
  cvt_f32_bf16_x4<<<1024, 256, 0, stream>>>(fc2_w, fc2_wb,
                                            N_EXP * D_MODEL * D_FF / 4);
  cvt_f32_bf16_x4<<<128, 256, 0, stream>>>(next_w, next_wb,
                                           D_MODEL * D_MODEL / 4);

  ln_to_bf16<<<N_TOK / 8, 256, 0, stream>>>(x, ln_g, ln_b, normed);

  dim3 blk(256);
  // x_attn = normed @ attn_w^T + attn_b + x          (mode 0)
  gemm_bf16_wmma<<<dim3(D_MODEL / 128, N_TOK / 128, 1), blk, 0, stream>>>(
      normed, attn_wb, attn_b, x, xattn, nullptr, D_MODEL, D_MODEL, N_TOK, 0, 0);
  // h = gelu(tok @ fc1_w[e]^T + fc1_b[e])            (mode 1, per expert)
  gemm_bf16_wmma<<<dim3(D_FF / 128, CHUNK_ROWS / 128, CHUNKS), blk, 0, stream>>>(
      xattn, fc1_wb, fc1_b, nullptr, hbuf, nullptr, D_MODEL, D_FF, CHUNK_ROWS,
      N_EXP - 1, 1);
  // y = h @ fc2_w[e]^T + fc2_b[e]                    (mode 2, per expert)
  gemm_bf16_wmma<<<dim3(D_MODEL / 128, CHUNK_ROWS / 128, CHUNKS), blk, 0,
                   stream>>>(hbuf, fc2_wb, fc2_b, nullptr, ybuf, nullptr, D_FF,
                             D_MODEL, CHUNK_ROWS, N_EXP - 1, 2);
  // out = gelu(y @ next_w^T + next_b) -> fp32        (mode 3)
  gemm_bf16_wmma<<<dim3(D_MODEL / 128, N_TOK / 128, 1), blk, 0, stream>>>(
      ybuf, next_wb, next_b, nullptr, nullptr, out, D_MODEL, D_MODEL, N_TOK, 0,
      3);
}
